// LSTMCRF_47768626266226
// MI455X (gfx1250) — matrix operand order repaired and checked by reading, not compiled, not run on hardware
//
#include <hip/hip_runtime.h>
#include <hip/hip_bf16.h>

// ---------------------------------------------------------------------------
// Problem constants (from reference)
// ---------------------------------------------------------------------------
#define BB    64
#define SS    1024
#define EE    512
#define HH    256          // hidden per direction
#define HD4   1024         // 4*H
#define NZ    2048         // both directions' 4H
#define KK    512          // input width (E == 2H == 512 for both layers)
#define TT    12
#define BS    (BB * SS)    // 65536
#define NEGV  (-10000.0f)

typedef __attribute__((ext_vector_type(16))) __bf16 bf16x16;
typedef __attribute__((ext_vector_type(8)))  __bf16 bf16x8;
typedef __attribute__((ext_vector_type(8)))  float  f32x8;
typedef int v4i_vec __attribute__((vector_size(16)));

// ---------------------------------------------------------------------------
// CDNA5 async global->LDS support (guarded; falls back to sync copy)
// ---------------------------------------------------------------------------
#if defined(__has_builtin)
#  if __has_builtin(__builtin_amdgcn_global_load_async_to_lds_b128)
#    define HAVE_ASYNC_COPY 1
#  else
#    define HAVE_ASYNC_COPY 0
#  endif
#  if __has_builtin(__builtin_amdgcn_s_wait_asynccnt)
#    define HAVE_WAIT_ASYNC 1
#  else
#    define HAVE_WAIT_ASYNC 0
#  endif
#else
#  define HAVE_ASYNC_COPY 0
#  define HAVE_WAIT_ASYNC 0
#endif

#define AS1P(p) ((__attribute__((address_space(1))) v4i_vec*)(p))
#define AS3P(p) ((__attribute__((address_space(3))) v4i_vec*)(p))

// copy 16 bytes global -> LDS (async on CDNA5 if available)
__device__ __forceinline__ void copy16_g2l(void* lds, const void* g) {
#if HAVE_ASYNC_COPY
    __builtin_amdgcn_global_load_async_to_lds_b128(AS1P(g), AS3P(lds), 0, 0);
#else
    *(uint4*)lds = *(const uint4*)g;
#endif
}

// wait for this wave's outstanding async LDS copies
__device__ __forceinline__ void wait_async0() {
#if HAVE_ASYNC_COPY
#  if HAVE_WAIT_ASYNC
    __builtin_amdgcn_s_wait_asynccnt(0);
#  else
    asm volatile("s_wait_asynccnt 0x0" ::: "memory");
#  endif
#endif
}

// ---------------------------------------------------------------------------
// Helpers
// ---------------------------------------------------------------------------
__device__ __forceinline__ __bf16 f2bf(float f) {
    unsigned u = __builtin_bit_cast(unsigned, f);
    unsigned r = (u + 0x7FFFu + ((u >> 16) & 1u)) >> 16;
    unsigned short s = (unsigned short)r;
    return __builtin_bit_cast(__bf16, s);
}

__device__ __forceinline__ float sigm(float x) { return 1.0f / (1.0f + __expf(-x)); }

// A fragment (16x32 bf16, MxK): lane holds caller-provided row, two 16B chunks
// at K = half*16 and K = half*16 + 8   (half = lane>>4)
__device__ __forceinline__ bf16x16 load_fragA(const __bf16* rowptr, int kk, int lane) {
    const __bf16* p = rowptr + kk + ((lane >> 4) << 4);
    bf16x8 lo = *(const bf16x8*)p;
    bf16x8 hi = *(const bf16x8*)(p + 8);
    return __builtin_shufflevector(lo, hi, 0, 1, 2, 3, 4, 5, 6, 7,
                                   8, 9, 10, 11, 12, 13, 14, 15);
}

// B fragment (32x16 bf16, KxN) built from row-major W (N x K):
// lane holds column n (= W row n), one contiguous 32B chunk at K = half*16
__device__ __forceinline__ bf16x16 load_fragB(const __bf16* rowptr, int kk, int lane) {
    return *(const bf16x16*)(rowptr + kk + ((lane >> 4) << 4));
}

__device__ __forceinline__ f32x8 wmma_bf16(bf16x16 a, bf16x16 b, f32x8 c) {
    return __builtin_amdgcn_wmma_f32_16x16x32_bf16(false, a, false, b, (short)0, c,
                                                   false, false);
}

// ---------------------------------------------------------------------------
// Kernel: lengths[b] = sum(sentences[b,:] > 0)
// ---------------------------------------------------------------------------
__global__ void k_lengths(const int* __restrict__ sent, int* __restrict__ len) {
    int b = threadIdx.x;
    if (b >= BB) return;
    int n = 0;
    for (int s = 0; s < SS; ++s) n += (sent[b * SS + s] > 0) ? 1 : 0;
    len[b] = n;
}

// ---------------------------------------------------------------------------
// Kernel: embedding gather + f32 -> bf16
// ---------------------------------------------------------------------------
__global__ void k_embed(const int* __restrict__ sent, const float* __restrict__ emb,
                        __bf16* __restrict__ X) {
    int row = blockIdx.x;                 // b*S + s
    int tok = sent[row];
    const float* e = emb + (size_t)tok * EE;
    __bf16* x = X + (size_t)row * EE;
    for (int i = threadIdx.x; i < EE; i += blockDim.x) x[i] = f2bf(e[i]);
}

// ---------------------------------------------------------------------------
// Kernel: per-layer weight conversion (f32 -> bf16) + bias fusion
// ---------------------------------------------------------------------------
__global__ void k_convw(const float* __restrict__ Wih, const float* __restrict__ Whh,
                        const float* __restrict__ bih, const float* __restrict__ bhh,
                        __bf16* __restrict__ Wbf, __bf16* __restrict__ Whhbf,
                        float* __restrict__ bcat, int l) {
    int tid = blockIdx.x * blockDim.x + threadIdx.x;
    int stride = gridDim.x * blockDim.x;
    const float* wih_l = Wih + (size_t)l * 2 * HD4 * KK;
    const float* whh_l = Whh + (size_t)l * 2 * HD4 * HH;
    for (int i = tid; i < 2 * HD4 * KK; i += stride) Wbf[i]   = f2bf(wih_l[i]);
    for (int i = tid; i < 2 * HD4 * HH; i += stride) Whhbf[i] = f2bf(whh_l[i]);
    for (int i = tid; i < 2 * HD4;      i += stride)
        bcat[i] = bih[(size_t)l * 2 * HD4 + i] + bhh[(size_t)l * 2 * HD4 + i];
}

// ---------------------------------------------------------------------------
// Kernel: FC weight conversion with zero pad 12 -> 16 rows
// ---------------------------------------------------------------------------
__global__ void k_convfc(const float* __restrict__ fcw, const float* __restrict__ fcb,
                         __bf16* __restrict__ fcwbf, float* __restrict__ fcbp) {
    int tid = blockIdx.x * blockDim.x + threadIdx.x;
    if (tid < 16 * KK) {
        int row = tid / KK, col = tid % KK;
        fcwbf[tid] = (row < TT) ? f2bf(fcw[row * KK + col]) : f2bf(0.0f);
    }
    if (tid < 16) fcbp[tid] = (tid < TT) ? fcb[tid] : 0.0f;
}

// ---------------------------------------------------------------------------
// Kernel: Z = X (M x 512, bf16) @ Wcat^T (2048 x 512, bf16) + bias  -> f32
// Block: 256 threads (8 waves). Block tile: 64 (M) x 128 (N). Grid (16, M/64).
// A tiles double-buffered in LDS via async global->LDS copies (ASYNCcnt);
// B streamed from global/L2 with prefetch hints.
// ---------------------------------------------------------------------------
#define LDSA_STRIDE 40
__global__ void k_gemm_z(const __bf16* __restrict__ X, const __bf16* __restrict__ W,
                         const float* __restrict__ bias, float* __restrict__ Z) {
    __shared__ __align__(16) __bf16 lds_x[2][64 * LDSA_STRIDE];

    const int lane = threadIdx.x & 31;
    const int wave = threadIdx.x >> 5;
    const int n0 = blockIdx.x * 128;
    const int m0 = blockIdx.y * 64;

    const int col  = n0 + wave * 16 + (lane & 15);
    const float bv = bias[col];
    f32x8 acc[4];
#pragma unroll
    for (int r = 0; r < 4; ++r)
#pragma unroll
        for (int e = 0; e < 8; ++e) acc[r][e] = bv;

    const __bf16* wrow = W + (size_t)col * KK;

    // per-thread staging slot: 16B of the 64x32 X tile
    const int srow = threadIdx.x >> 2;
    const int schk = (threadIdx.x & 3) << 3;

    // prologue: stage chunk 0 into buffer 0
    copy16_g2l(&lds_x[0][srow * LDSA_STRIDE + schk],
               &X[(size_t)(m0 + srow) * KK + 0 + schk]);

    for (int kk = 0, it = 0; kk < KK; kk += 32, ++it) {
        const int cur = it & 1;

        wait_async0();        // this wave's async copies have landed
        __syncthreads();      // everyone's copies landed; prev buffer free

        if (kk + 32 < KK) {   // stage next chunk into the other buffer
            copy16_g2l(&lds_x[cur ^ 1][srow * LDSA_STRIDE + schk],
                       &X[(size_t)(m0 + srow) * KK + (kk + 32) + schk]);
            __builtin_prefetch(wrow + kk + 32, 0, 1);   // next B chunk -> L2/L0
        }

        bf16x16 bfrag = load_fragB(wrow, kk, lane);
#pragma unroll
        for (int r = 0; r < 4; ++r) {
            bf16x16 afrag =
                load_fragA(&lds_x[cur][(r * 16 + (lane & 15)) * LDSA_STRIDE], 0, lane);
            acc[r] = wmma_bf16(afrag, bfrag, acc[r]);
        }
    }

#pragma unroll
    for (int r = 0; r < 4; ++r)
#pragma unroll
        for (int e = 0; e < 8; ++e) {
            int row = m0 + r * 16 + ((lane >> 4) << 3) + e;
            Z[(size_t)row * NZ + col] = acc[r][e];
        }
}

// ---------------------------------------------------------------------------
// Kernel: LSTM recurrence scan. Grid = 2 blocks (dir), 512 threads (16 waves).
// h (64x256 bf16) lives in LDS; c and h persist in VGPRs per wave-task.
// Per step: gates = Z[row(t)] + h @ Whh^T (WMMA, K=256), fused gate math,
// masked state update, reversed output scatter for the backward direction.
// ---------------------------------------------------------------------------
#define HSTRIDE 264   // 256 + 8 pad elems (16B) to break LDS bank conflicts
__global__ void k_scan(const float* __restrict__ Z, const __bf16* __restrict__ Whhbf,
                       const int* __restrict__ len, __bf16* __restrict__ Xout) {
    __shared__ __align__(16) __bf16 h_lds[64 * HSTRIDE];
    __shared__ int len_s[BB];

    const int dir  = blockIdx.x;
    const int lane = threadIdx.x & 31;
    const int wave = threadIdx.x >> 5;

    for (int i = threadIdx.x; i < 64 * HSTRIDE; i += blockDim.x)
        h_lds[i] = f2bf(0.0f);
    if (threadIdx.x < BB) len_s[threadIdx.x] = len[threadIdx.x];
    __syncthreads();

    float c_reg[4][8];
    float h_reg[4][8];
#pragma unroll
    for (int j = 0; j < 4; ++j)
#pragma unroll
        for (int e = 0; e < 8; ++e) { c_reg[j][e] = 0.0f; h_reg[j][e] = 0.0f; }

    for (int t = 0; t < SS; ++t) {
#pragma unroll
        for (int task = 0; task < 4; ++task) {
            const int id = wave * 4 + task;
            const int mt = id >> 4;            // batch tile (0..3)
            const int u0 = (id & 15) << 4;     // hidden-unit tile base (0..240)
            const int nloc = lane & 15;

            // init accumulators from precomputed x@Wih^T + bias (f32)
            f32x8 acc[4];
#pragma unroll
            for (int g = 0; g < 4; ++g) {
                const int col = dir * HD4 + g * HH + u0 + nloc;
#pragma unroll
                for (int e = 0; e < 8; ++e) {
                    const int b  = mt * 16 + ((lane >> 4) << 3) + e;
                    const int lb = len_s[b];
                    const int tx = (dir == 0) ? t : ((t < lb) ? lb - 1 - t : t);
                    acc[g][e] = Z[((size_t)b * SS + tx) * NZ + col];
                }
            }

            // K loop over hidden state (K = 256); Whh rows stream from L2
            for (int kk = 0; kk < HH; kk += 32) {
                bf16x16 afrag =
                    load_fragA(&h_lds[(mt * 16 + nloc) * HSTRIDE], kk, lane);
#pragma unroll
                for (int g = 0; g < 4; ++g) {
                    const __bf16* wrow =
                        Whhbf + (size_t)(dir * HD4 + g * HH + u0 + nloc) * HH;
                    if (kk + 32 < HH) __builtin_prefetch(wrow + kk + 32, 0, 1);
                    acc[g] = wmma_bf16(afrag, load_fragB(wrow, kk, lane), acc[g]);
                }
            }

            // gate fusion + masked state update + output scatter
#pragma unroll
            for (int e = 0; e < 8; ++e) {
                const int b   = mt * 16 + ((lane >> 4) << 3) + e;
                const int lb  = len_s[b];
                const bool m  = (t < lb);
                const float iv = sigm(acc[0][e]);
                const float fv = sigm(acc[1][e]);
                const float gv = tanhf(acc[2][e]);
                const float ov = sigm(acc[3][e]);
                const float cn = fv * c_reg[task][e] + iv * gv;
                const float hn = ov * tanhf(cn);
                if (m) { c_reg[task][e] = cn; h_reg[task][e] = hn; }
                const float ys = m ? hn : 0.0f;
                const int q = (dir == 0) ? t : ((t < lb) ? lb - 1 - t : t);
                Xout[((size_t)b * SS + q) * (2 * HH) + dir * HH + u0 + nloc] =
                    f2bf(ys);
            }
        }
        __syncthreads();
        // publish masked h for next step
#pragma unroll
        for (int task = 0; task < 4; ++task) {
            const int id = wave * 4 + task;
            const int mt = id >> 4;
            const int u0 = (id & 15) << 4;
            const int nloc = lane & 15;
#pragma unroll
            for (int e = 0; e < 8; ++e) {
                const int b = mt * 16 + ((lane >> 4) << 3) + e;
                h_lds[b * HSTRIDE + u0 + nloc] = f2bf(h_reg[task][e]);
            }
        }
        __syncthreads();
    }
}

// ---------------------------------------------------------------------------
// Kernel: feats = Y (BS x 512, bf16) @ fc_w^T (16 x 512, padded) + fc_b
// ---------------------------------------------------------------------------
__global__ void k_fc(const __bf16* __restrict__ Y, const __bf16* __restrict__ Wf,
                     const float* __restrict__ bf, float* __restrict__ feats) {
    const int lane = threadIdx.x & 31;
    const int wave = threadIdx.x >> 5;
    const int r0 = (blockIdx.x * 8 + wave) * 16;
    const int nloc = lane & 15;

    f32x8 acc;
    const float bv = bf[nloc];
#pragma unroll
    for (int e = 0; e < 8; ++e) acc[e] = bv;

    const __bf16* wrow = Wf + (size_t)nloc * KK;
    for (int kk = 0; kk < KK; kk += 32) {
        bf16x16 afrag = load_fragA(Y + (size_t)(r0 + nloc) * KK, kk, lane);
        acc = wmma_bf16(afrag, load_fragB(wrow, kk, lane), acc);
    }
#pragma unroll
    for (int e = 0; e < 8; ++e) {
        int row = r0 + ((lane >> 4) << 3) + e;
        feats[(size_t)row * 16 + nloc] = acc[e];
    }
}

// ---------------------------------------------------------------------------
// Kernel: Viterbi decode. One thread per batch element.
// out[0..63] = max scores; out[64 + b*S + t] = tag (as float).
// ---------------------------------------------------------------------------
__global__ void k_viterbi(const float* __restrict__ feats, const float* __restrict__ trans,
                          const int* __restrict__ len, int* __restrict__ bp,
                          float* __restrict__ out) {
    __shared__ float tr[TT * TT];
    for (int i = threadIdx.x; i < TT * TT; i += blockDim.x) tr[i] = trans[i];
    __syncthreads();

    const int b = threadIdx.x;
    if (b >= BB) return;
    const int lb = len[b];

    float fs[TT];
#pragma unroll
    for (int j = 0; j < TT; ++j) fs[j] = (j == 0) ? 0.0f : NEGV;  // START=0

    for (int t = 0; t < SS; ++t) {
        const bool m = (t < lb);
        const float* em = feats + ((size_t)b * SS + t) * 16;
        float nf[TT];
        int* bpt = bp + ((size_t)b * SS + t) * TT;
#pragma unroll
        for (int j = 0; j < TT; ++j) {
            float best = fs[0] + tr[j * TT + 0];
            int arg = 0;
#pragma unroll
            for (int k = 1; k < TT; ++k) {
                float v = fs[k] + tr[j * TT + k];
                if (v > best) { best = v; arg = k; }
            }
            nf[j] = best + em[j];
            bpt[j] = arg;
        }
        if (m)
#pragma unroll
            for (int j = 0; j < TT; ++j) fs[j] = nf[j];
    }

    // terminal transition to END (tag 4)
    float best = fs[0] + tr[4 * TT + 0];
    int nt = 0;
#pragma unroll
    for (int j = 1; j < TT; ++j) {
        float v = fs[j] + tr[4 * TT + j];
        if (v > best) { best = v; nt = j; }
    }
    out[b] = best;

    float* tag_out = out + BB + (size_t)b * SS;
    for (int t = SS - 1; t >= 1; --t) {
        const bool m = (t < lb);
        const int tag_i = m ? nt : 4;  // END when masked
        const int ntn = bp[((size_t)b * SS + t) * TT + nt];
        if (m) nt = ntn;
        tag_out[t] = (float)tag_i;
    }
    tag_out[0] = (float)nt;
}

// ---------------------------------------------------------------------------
// Host launch
// ---------------------------------------------------------------------------
extern "C" void kernel_launch(void* const* d_in, const int* in_sizes, int n_in,
                              void* d_out, int out_size, void* d_ws, size_t ws_size,
                              hipStream_t stream) {
    const int*   sent  = (const int*)d_in[0];
    const float* emb   = (const float*)d_in[1];
    const float* Wih   = (const float*)d_in[2];
    const float* Whh   = (const float*)d_in[3];
    const float* bih   = (const float*)d_in[4];
    const float* bhh   = (const float*)d_in[5];
    const float* fcw   = (const float*)d_in[6];
    const float* fcbv  = (const float*)d_in[7];
    const float* trans = (const float*)d_in[8];
    float* out = (float*)d_out;

    char* ws = (char*)d_ws;
    size_t off = 0;
    auto carve = [&](size_t bytes) -> char* {
        char* p = ws + off;
        off = (off + bytes + 255) & ~(size_t)255;
        return p;
    };
    int*    len    = (int*)carve(BB * sizeof(int));
    __bf16* X0     = (__bf16*)carve((size_t)BS * KK * 2);
    __bf16* X1     = (__bf16*)carve((size_t)BS * KK * 2);
    float*  Z      = (float*)carve((size_t)BS * NZ * 4);
    __bf16* Wbf    = (__bf16*)carve((size_t)NZ * KK * 2);
    __bf16* Whhbf  = (__bf16*)carve((size_t)NZ * HH * 2);
    float*  bcat   = (float*)carve(NZ * 4);
    __bf16* fcwbf  = (__bf16*)carve((size_t)16 * KK * 2);
    float*  fcbp   = (float*)carve(16 * 4);
    float*  feats  = (float*)carve((size_t)BS * 16 * 4);
    int*    bp     = (int*)carve((size_t)BS * TT * 4);
    (void)ws_size; (void)in_sizes; (void)n_in; (void)out_size;

    k_lengths<<<1, 64, 0, stream>>>(sent, len);
    k_embed<<<BS, 256, 0, stream>>>(sent, emb, X0);
    k_convfc<<<(16 * KK + 255) / 256, 256, 0, stream>>>(fcw, fcbv, fcwbf, fcbp);

    __bf16* Xin = X0;
    __bf16* Xo  = X1;
    for (int l = 0; l < 2; ++l) {
        k_convw<<<2048, 256, 0, stream>>>(Wih, Whh, bih, bhh, Wbf, Whhbf, bcat, l);
        dim3 gz(NZ / 128, BS / 64);
        k_gemm_z<<<gz, 256, 0, stream>>>(Xin, Wbf, bcat, Z);
        k_scan<<<2, 512, 0, stream>>>(Z, Whhbf, len, Xo);
        __bf16* tmp = Xin; Xin = Xo; Xo = tmp;
    }

    k_fc<<<BS / 128, 256, 0, stream>>>(Xin, fcwbf, fcbp, feats);
    k_viterbi<<<1, 64, 0, stream>>>(feats, trans, len, bp, out);
}
